// GnBlock_25469156065752
// MI455X (gfx1250) — matrix operand order repaired and checked by reading, not compiled, not run on hardware
//
#include <hip/hip_runtime.h>
#include <cstdint>
#include <cstddef>

// ---------------------------------------------------------------------------
// CDNA5 (gfx1250) GraphNet block: edge MLP+LN -> scatter-add -> node MLP+LN.
// bf16 WMMA 16x16x32, f32 accumulate. Weights pre-swizzled into LDS.
// Each wave processes TWO 16-row tiles so every B-fragment LDS read feeds
// two WMMAs; k-loops are partially unrolled to keep VGPR pressure bounded.
// ---------------------------------------------------------------------------

typedef __attribute__((ext_vector_type(16))) __bf16        v16bf;
typedef __attribute__((ext_vector_type(8)))  float         v8f;
typedef __attribute__((ext_vector_type(4)))  float         f32x4;
typedef __attribute__((ext_vector_type(4)))  unsigned int  u32x4;

union Frag {
    v16bf        v;
    u32x4        q[2];
    unsigned int u[8];
};

#define WAVES 4
#define HSTRIDE 132  // 128 + pad: keeps 16B alignment, staggers LDS banks by row

// Edge-kernel dynamic-LDS layout (bytes)
constexpr int EW1   = 0;                 // 12 kb * 8 nt * 1024B = 98304
constexpr int EW2   = 98304;             // 32768
constexpr int EW3   = 131072;
constexpr int EW4   = 163840;
constexpr int EBIAS = 196608;            // b1,b2,b3,b4,g,be : 6*128*4 = 3072
constexpr int EHST  = 199680;            // 4 waves * 2 tiles * 16*132*4 = 67584
constexpr int EDGE_SMEM = 267264;
// Node-kernel dynamic-LDS layout (bytes)
constexpr int NW1   = 0;                 // 8 kb -> 65536
constexpr int NW2   = 65536;
constexpr int NW3   = 98304;
constexpr int NW4   = 131072;
constexpr int NBIAS = 163840;
constexpr int NHST  = 166912;
constexpr int NODE_SMEM = 234496;

// f32 pair -> packed bf16 dword, round-to-nearest (ties away from zero).
// 2x v_add_u32 + 1x v_perm_b32 (selects the high halves of pa|pb).
__device__ __forceinline__ unsigned int pack_bf16(float a, float b) {
    const unsigned int pa = __float_as_uint(a) + 0x8000u;
    const unsigned int pb = __float_as_uint(b) + 0x8000u;
    return __builtin_amdgcn_perm(pb, pa, 0x07060302u);
}

// A-fragment (16x32 bf16, MxK). Lane layout per ISA 7.12.2:
// lanes 0-15: VGPR0-3 K=0..7, VGPR4-7 K=16..23 ; lanes 16-31: +8.
// p must already include the per-half +8-float offset; 16B aligned.
__device__ __forceinline__ void load_A_f32(Frag& a, const float* p) {
    f32x4 q0 = *(const f32x4*)(p + 0);
    f32x4 q1 = *(const f32x4*)(p + 4);
    f32x4 q2 = *(const f32x4*)(p + 16);
    f32x4 q3 = *(const f32x4*)(p + 20);
    a.u[0] = pack_bf16(q0.x, q0.y);
    a.u[1] = pack_bf16(q0.z, q0.w);
    a.u[2] = pack_bf16(q1.x, q1.y);
    a.u[3] = pack_bf16(q1.z, q1.w);
    a.u[4] = pack_bf16(q2.x, q2.y);
    a.u[5] = pack_bf16(q2.z, q2.w);
    a.u[6] = pack_bf16(q3.x, q3.y);
    a.u[7] = pack_bf16(q3.z, q3.w);
}

// One K-block (K=32) of two M-tiles against 8 N-tiles of pre-swizzled
// LDS weights; each B fragment is read once and used by two WMMAs.
__device__ __forceinline__ void wmma_kblock2(v8f acc0[8], v8f acc1[8],
                                             const Frag& a0, const Frag& a1,
                                             const unsigned int* wl, int kb, int lane) {
#pragma unroll
    for (int nt = 0; nt < 8; ++nt) {
        Frag b;
        const u32x4* q = (const u32x4*)(wl + ((kb * 8 + nt) << 8) + (lane << 3));
        b.q[0] = q[0];
        b.q[1] = q[1];
        acc0[nt] = __builtin_amdgcn_wmma_f32_16x16x32_bf16(
            false, a0.v, false, b.v, (short)0, acc0[nt], false, false);
        acc1[nt] = __builtin_amdgcn_wmma_f32_16x16x32_bf16(
            false, a1.v, false, b.v, (short)0, acc1[nt], false, false);
    }
}

__device__ __forceinline__ void zero_acc(v8f acc[8]) {
    const v8f z = {0.f, 0.f, 0.f, 0.f, 0.f, 0.f, 0.f, 0.f};
#pragma unroll
    for (int nt = 0; nt < 8; ++nt) acc[nt] = z;
}

// Hidden layer for two tiles: A from per-wave staging, K=128 (4 k-blocks).
// unroll 2 keeps the in-flight conversion window (VGPR pressure) bounded.
__device__ __forceinline__ void hidden_layer2(v8f acc0[8], v8f acc1[8],
                                              const unsigned int* wl,
                                              const float* hst0, const float* hst1,
                                              int lane) {
    const int M  = lane & 15;
    const int hi = (lane >> 4) << 3;
#pragma unroll 2
    for (int kb = 0; kb < 4; ++kb) {
        Frag a0, a1;
        load_A_f32(a0, hst0 + M * HSTRIDE + kb * 32 + hi);
        load_A_f32(a1, hst1 + M * HSTRIDE + kb * 32 + hi);
        wmma_kblock2(acc0, acc1, a0, a1, wl, kb, lane);
    }
}

// D layout (f32 16x16): lanes 0-15 -> N=lane, VGPR r -> M=r ; lanes 16-31 -> M=r+8.
__device__ __forceinline__ void store_h(const v8f acc[8], float* hst,
                                        const float* bias, int lane, bool relu) {
    const int col0 = lane & 15;
    const int rb   = (lane >> 4) << 3;
#pragma unroll
    for (int nt = 0; nt < 8; ++nt) {
        const int col = nt * 16 + col0;
        const float bb = bias[col];
#pragma unroll
        for (int r = 0; r < 8; ++r) {
            float v = acc[nt][r] + bb;
            if (relu) v = fmaxf(v, 0.0f);
            hst[(rb + r) * HSTRIDE + col] = v;
        }
    }
}

// Pre-swizzle a [K=32*kblocks, 128] row-major f32 weight matrix into bf16
// B-fragments.  Fragment dword d: lane = d>>3, v = d&7 ; B layout per ISA:
// lanes 0-15: N=lane, VGPR0-3 K=0..7, VGPR4-7 K=16..23 ; lanes 16-31: N=lane-16, K += 8.
__device__ __forceinline__ void fill_w(unsigned int* dst, const float* __restrict__ w,
                                       int kblocks, int tid, int nthr) {
    const int total = kblocks * 2048;  // dwords (8 frags * 256 per k-block)
    for (int t = tid; t < total; t += nthr) {
        const int fragIdx = t >> 8;
        const int d    = t & 255;
        const int lane = d >> 3;
        const int v    = d & 7;
        const int nt   = fragIdx & 7;
        const int kb   = fragIdx >> 3;
        const int n    = nt * 16 + (lane & 15);
        const int k    = kb * 32 + ((lane & 16) ? 8 : 0) + ((v & 4) ? 8 : 0) + (v << 1);
        dst[t] = pack_bf16(w[k * 128 + n], w[(k + 1) * 128 + n]);
    }
}

// LayerNorm stats for one staged tile: lane handles row (lane&15), half (lane>>4).
__device__ __forceinline__ void ln_stats(const float* hr, float& mu, float& rs) {
    f32x4 vs0 = {0.f, 0.f, 0.f, 0.f};
    f32x4 vs1 = {0.f, 0.f, 0.f, 0.f};
#pragma unroll 4
    for (int k = 0; k < 64; k += 4) {
        const f32x4 v = *(const f32x4*)(hr + k);
        vs0 += v;
        vs1 += v * v;
    }
    float s0 = vs0.x + vs0.y + vs0.z + vs0.w;
    float s1 = vs1.x + vs1.y + vs1.z + vs1.w;
    s0 += __shfl_xor(s0, 16, 32);
    s1 += __shfl_xor(s1, 16, 32);
    mu = s0 * (1.0f / 128.0f);
    const float var = s1 * (1.0f / 128.0f) - mu * mu;
    rs = rsqrtf(var + 1e-5f);
}

// ---------------------------------------------------------------------------
__global__ void __launch_bounds__(128)
gn_edge_kernel(const float* __restrict__ x, const float* __restrict__ edge_attr,
               const int* __restrict__ edge_index,
               const float* __restrict__ w1, const float* __restrict__ b1,
               const float* __restrict__ w2, const float* __restrict__ b2,
               const float* __restrict__ w3, const float* __restrict__ b3,
               const float* __restrict__ w4, const float* __restrict__ b4,
               const float* __restrict__ g,  const float* __restrict__ be,
               float* __restrict__ edge_out, float* __restrict__ agg, int E)
{
    extern __shared__ __align__(16) char smem[];
    const int tid = threadIdx.x;
    unsigned int* wl1 = (unsigned int*)(smem + EW1);
    unsigned int* wl2 = (unsigned int*)(smem + EW2);
    unsigned int* wl3 = (unsigned int*)(smem + EW3);
    unsigned int* wl4 = (unsigned int*)(smem + EW4);
    float* bl = (float*)(smem + EBIAS);

    fill_w(wl1, w1, 12, tid, blockDim.x);
    fill_w(wl2, w2, 4, tid, blockDim.x);
    fill_w(wl3, w3, 4, tid, blockDim.x);
    fill_w(wl4, w4, 4, tid, blockDim.x);
    for (int i = tid; i < 128; i += blockDim.x) {
        bl[i] = b1[i]; bl[128 + i] = b2[i]; bl[256 + i] = b3[i];
        bl[384 + i] = b4[i]; bl[512 + i] = g[i]; bl[640 + i] = be[i];
    }
    __syncthreads();

    const int wave = tid >> 5;
    const int lane = tid & 31;
    float* hst0 = (float*)(smem + EHST) + wave * (32 * HSTRIDE);
    float* hst1 = hst0 + 16 * HSTRIDE;
    const int M  = lane & 15;
    const int hi = (lane >> 4) << 3;
    const int numTiles = E >> 4;           // 37500 (even)
    const int numPairs = (numTiles + 1) >> 1;
    const int slot   = blockIdx.x * WAVES + wave;
    const int nslots = gridDim.x * WAVES;

    for (int pair = slot; pair < numPairs; pair += nslots) {
        const int t0 = pair * 2;
        const int t1 = t0 + 1;
        const bool v1 = (t1 < numTiles);
        const int tv1 = v1 ? t1 : t0;

        const int e0 = (t0 << 4) + M;
        const int e1 = (tv1 << 4) + M;
        const int snd0 = edge_index[e0];
        const int rcv0 = edge_index[E + e0];
        const int snd1 = edge_index[e1];
        const int rcv1 = edge_index[E + e1];

        // Layer 1: [16 x 384] @ [384 x 128], A gathered from global (x2 tiles)
        v8f acc0[8], acc1[8];
        zero_acc(acc0);
        zero_acc(acc1);
#pragma unroll 4
        for (int kb = 0; kb < 12; ++kb) {
            const float* base0 =
                  (kb < 4) ? (x + (size_t)snd0 * 128 + kb * 32)
                : (kb < 8) ? (x + (size_t)rcv0 * 128 + (kb - 4) * 32)
                           : (edge_attr + (size_t)e0 * 128 + (kb - 8) * 32);
            const float* base1 =
                  (kb < 4) ? (x + (size_t)snd1 * 128 + kb * 32)
                : (kb < 8) ? (x + (size_t)rcv1 * 128 + (kb - 4) * 32)
                           : (edge_attr + (size_t)e1 * 128 + (kb - 8) * 32);
            Frag a0, a1;
            load_A_f32(a0, base0 + hi);
            load_A_f32(a1, base1 + hi);
            wmma_kblock2(acc0, acc1, a0, a1, wl1, kb, lane);
        }
        store_h(acc0, hst0, bl + 0, lane, true);
        store_h(acc1, hst1, bl + 0, lane, true);

        zero_acc(acc0); zero_acc(acc1);
        hidden_layer2(acc0, acc1, wl2, hst0, hst1, lane);
        store_h(acc0, hst0, bl + 128, lane, true);
        store_h(acc1, hst1, bl + 128, lane, true);

        zero_acc(acc0); zero_acc(acc1);
        hidden_layer2(acc0, acc1, wl3, hst0, hst1, lane);
        store_h(acc0, hst0, bl + 256, lane, true);
        store_h(acc1, hst1, bl + 256, lane, true);

        zero_acc(acc0); zero_acc(acc1);
        hidden_layer2(acc0, acc1, wl4, hst0, hst1, lane);
        store_h(acc0, hst0, bl + 384, lane, false);
        store_h(acc1, hst1, bl + 384, lane, false);

        // LayerNorm + residual + scatter-add, per tile.
        const int half = lane >> 4;
#pragma unroll
        for (int tt = 0; tt < 2; ++tt) {
            if (tt == 1 && !v1) break;
            const float* hst = (tt == 0) ? hst0 : hst1;
            const int e   = (tt == 0) ? e0 : e1;
            const int rcv = (tt == 0) ? rcv0 : rcv1;
            const float* hr = hst + M * HSTRIDE + half * 64;
            float mu, rs;
            ln_stats(hr, mu, rs);

            const float* ea = edge_attr + (size_t)e * 128 + half * 64;
            float*       eo = edge_out  + (size_t)e * 128 + half * 64;
            float*       ag = agg + (size_t)rcv * 128 + half * 64;
            const float* gp = bl + 512 + half * 64;
            const float* bp = bl + 640 + half * 64;
#pragma unroll 4
            for (int k = 0; k < 64; k += 4) {
                const f32x4 h4 = *(const f32x4*)(hr + k);
                const f32x4 g4 = *(const f32x4*)(gp + k);
                const f32x4 p4 = *(const f32x4*)(bp + k);
                const f32x4 v4 = (h4 - mu) * rs * g4 + p4;
                const f32x4 e4 = *(const f32x4*)(ea + k);
                *(f32x4*)(eo + k) = e4 + v4;          // residual edge output
                atomicAdd(&ag[k + 0], v4.x);           // segment_sum to receiver
                atomicAdd(&ag[k + 1], v4.y);
                atomicAdd(&ag[k + 2], v4.z);
                atomicAdd(&ag[k + 3], v4.w);
            }
        }
    }
}

// ---------------------------------------------------------------------------
__global__ void __launch_bounds__(128)
gn_node_kernel(const float* __restrict__ x, const float* __restrict__ agg,
               const float* __restrict__ w1, const float* __restrict__ b1,
               const float* __restrict__ w2, const float* __restrict__ b2,
               const float* __restrict__ w3, const float* __restrict__ b3,
               const float* __restrict__ w4, const float* __restrict__ b4,
               const float* __restrict__ g,  const float* __restrict__ be,
               float* __restrict__ out_x, int N)
{
    extern __shared__ __align__(16) char smem[];
    const int tid = threadIdx.x;
    unsigned int* wl1 = (unsigned int*)(smem + NW1);
    unsigned int* wl2 = (unsigned int*)(smem + NW2);
    unsigned int* wl3 = (unsigned int*)(smem + NW3);
    unsigned int* wl4 = (unsigned int*)(smem + NW4);
    float* bl = (float*)(smem + NBIAS);

    fill_w(wl1, w1, 8, tid, blockDim.x);
    fill_w(wl2, w2, 4, tid, blockDim.x);
    fill_w(wl3, w3, 4, tid, blockDim.x);
    fill_w(wl4, w4, 4, tid, blockDim.x);
    for (int i = tid; i < 128; i += blockDim.x) {
        bl[i] = b1[i]; bl[128 + i] = b2[i]; bl[256 + i] = b3[i];
        bl[384 + i] = b4[i]; bl[512 + i] = g[i]; bl[640 + i] = be[i];
    }
    __syncthreads();

    const int wave = tid >> 5;
    const int lane = tid & 31;
    float* hst0 = (float*)(smem + NHST) + wave * (32 * HSTRIDE);
    float* hst1 = hst0 + 16 * HSTRIDE;
    const int M  = lane & 15;
    const int hi = (lane >> 4) << 3;
    const int numTiles = N >> 4;           // 3125 (odd)
    const int numPairs = (numTiles + 1) >> 1;
    const int slot   = blockIdx.x * WAVES + wave;
    const int nslots = gridDim.x * WAVES;

    for (int pair = slot; pair < numPairs; pair += nslots) {
        const int t0 = pair * 2;
        const int t1 = t0 + 1;
        const bool v1 = (t1 < numTiles);
        const int tv1 = v1 ? t1 : t0;

        const int node0 = (t0 << 4) + M;
        const int node1 = (tv1 << 4) + M;

        // Layer 1: [16 x 256] @ [256 x 128], A = concat(x, agg) (x2 tiles)
        v8f acc0[8], acc1[8];
        zero_acc(acc0);
        zero_acc(acc1);
#pragma unroll 4
        for (int kb = 0; kb < 8; ++kb) {
            const float* base0 = (kb < 4)
                ? (x   + (size_t)node0 * 128 + kb * 32)
                : (agg + (size_t)node0 * 128 + (kb - 4) * 32);
            const float* base1 = (kb < 4)
                ? (x   + (size_t)node1 * 128 + kb * 32)
                : (agg + (size_t)node1 * 128 + (kb - 4) * 32);
            Frag a0, a1;
            load_A_f32(a0, base0 + hi);
            load_A_f32(a1, base1 + hi);
            wmma_kblock2(acc0, acc1, a0, a1, wl1, kb, lane);
        }
        store_h(acc0, hst0, bl + 0, lane, true);
        store_h(acc1, hst1, bl + 0, lane, true);

        zero_acc(acc0); zero_acc(acc1);
        hidden_layer2(acc0, acc1, wl2, hst0, hst1, lane);
        store_h(acc0, hst0, bl + 128, lane, true);
        store_h(acc1, hst1, bl + 128, lane, true);

        zero_acc(acc0); zero_acc(acc1);
        hidden_layer2(acc0, acc1, wl3, hst0, hst1, lane);
        store_h(acc0, hst0, bl + 256, lane, true);
        store_h(acc1, hst1, bl + 256, lane, true);

        zero_acc(acc0); zero_acc(acc1);
        hidden_layer2(acc0, acc1, wl4, hst0, hst1, lane);
        store_h(acc0, hst0, bl + 384, lane, false);
        store_h(acc1, hst1, bl + 384, lane, false);

        const int half = lane >> 4;
#pragma unroll
        for (int tt = 0; tt < 2; ++tt) {
            if (tt == 1 && !v1) break;
            const float* hst = (tt == 0) ? hst0 : hst1;
            const int node = (tt == 0) ? node0 : node1;
            const float* hr = hst + M * HSTRIDE + half * 64;
            float mu, rs;
            ln_stats(hr, mu, rs);

            const float* xr = x     + (size_t)node * 128 + half * 64;
            float*       xo = out_x + (size_t)node * 128 + half * 64;
            const float* gp = bl + 512 + half * 64;
            const float* bp = bl + 640 + half * 64;
#pragma unroll 4
            for (int k = 0; k < 64; k += 4) {
                const f32x4 h4 = *(const f32x4*)(hr + k);
                const f32x4 g4 = *(const f32x4*)(gp + k);
                const f32x4 p4 = *(const f32x4*)(bp + k);
                const f32x4 v4 = (h4 - mu) * rs * g4 + p4;
                const f32x4 x4 = *(const f32x4*)(xr + k);
                *(f32x4*)(xo + k) = x4 + v4;          // residual node output
            }
        }
    }
}

// ---------------------------------------------------------------------------
extern "C" void kernel_launch(void* const* d_in, const int* in_sizes, int n_in,
                              void* d_out, int out_size, void* d_ws, size_t ws_size,
                              hipStream_t stream) {
    const float* x         = (const float*)d_in[0];
    const float* edge_attr = (const float*)d_in[1];
    const int*   edge_index= (const int*)d_in[2];
    const float* e_w1 = (const float*)d_in[3];  const float* e_b1 = (const float*)d_in[4];
    const float* e_w2 = (const float*)d_in[5];  const float* e_b2 = (const float*)d_in[6];
    const float* e_w3 = (const float*)d_in[7];  const float* e_b3 = (const float*)d_in[8];
    const float* e_w4 = (const float*)d_in[9];  const float* e_b4 = (const float*)d_in[10];
    const float* e_g  = (const float*)d_in[11]; const float* e_be = (const float*)d_in[12];
    const float* n_w1 = (const float*)d_in[13]; const float* n_b1 = (const float*)d_in[14];
    const float* n_w2 = (const float*)d_in[15]; const float* n_b2 = (const float*)d_in[16];
    const float* n_w3 = (const float*)d_in[17]; const float* n_b3 = (const float*)d_in[18];
    const float* n_w4 = (const float*)d_in[19]; const float* n_b4 = (const float*)d_in[20];
    const float* n_g  = (const float*)d_in[21]; const float* n_be = (const float*)d_in[22];

    const int Hf = 128;
    const int N = in_sizes[0] / Hf;
    const int E = in_sizes[1] / Hf;

    float* out_x = (float*)d_out;                       // [N,128]
    float* out_e = out_x + (size_t)N * Hf;              // [E,128]
    float* agg   = (float*)d_ws;                        // [N,128] scratch

    hipMemsetAsync(agg, 0, (size_t)N * Hf * sizeof(float), stream);

    gn_edge_kernel<<<1024, 128, EDGE_SMEM, stream>>>(
        x, edge_attr, edge_index,
        e_w1, e_b1, e_w2, e_b2, e_w3, e_b3, e_w4, e_b4, e_g, e_be,
        out_e, agg, E);

    gn_node_kernel<<<512, 128, NODE_SMEM, stream>>>(
        x, agg,
        n_w1, n_b1, n_w2, n_b2, n_w3, n_b3, n_w4, n_b4, n_g, n_be,
        out_x, N);
}
